// Net_80582176408381
// MI455X (gfx1250) — matrix operand order repaired and checked by reading, not compiled, not run on hardware
//
#include <hip/hip_runtime.h>
#include <hip/hip_bf16.h>

#define NEG_SLOPE 0.2f

typedef __bf16 bf16_t;
typedef __attribute__((ext_vector_type(16))) __bf16 v16bf;
typedef __attribute__((ext_vector_type(8)))  __bf16 v8bf;
typedef __attribute__((ext_vector_type(8)))  float  v8f;

// ---------------------------------------------------------------- elementwise
__global__ void k_cast_bf16(const float* __restrict__ s, bf16_t* __restrict__ d, size_t n) {
  size_t i = (size_t)blockIdx.x * blockDim.x + threadIdx.x;
  size_t st = (size_t)gridDim.x * blockDim.x;
  for (; i < n; i += st) d[i] = (bf16_t)s[i];
}

__global__ void k_fill_f32(float* __restrict__ p, float v, size_t n) {
  size_t i = (size_t)blockIdx.x * blockDim.x + threadIdx.x;
  size_t st = (size_t)gridDim.x * blockDim.x;
  for (; i < n; i += st) p[i] = v;
}

// wt[n*K + k] = w[k*Nout + n]  (store weights K-major for contiguous B fragments)
__global__ void k_transpose_bf16(const float* __restrict__ w, bf16_t* __restrict__ wt,
                                 int K, int Nout) {
  size_t total = (size_t)K * Nout;
  size_t i = (size_t)blockIdx.x * blockDim.x + threadIdx.x;
  size_t st = (size_t)gridDim.x * blockDim.x;
  for (; i < total; i += st) {
    int n = (int)(i / K);
    int k = (int)(i - (size_t)n * K);
    wt[i] = (bf16_t)w[(size_t)k * Nout + n];
  }
}

// ------------------------------------------------------------------ WMMA GEMM
// C[M,Nout] = A[M,K] (bf16, row-major) x Bt[Nout,K] (bf16, K-major).
// One wave computes a 16 x (16*NT) strip: the A fragment is loaded once per
// 32-wide K chunk and reused across NT v_wmma_f32_16x16x32_bf16 issues.
template <int NT>
__global__ void k_gemm_bf16_wmma(const bf16_t* __restrict__ A,
                                 const bf16_t* __restrict__ Bt,
                                 float* __restrict__ C,
                                 int M, int K, int Nout, int tilesTotal) {
  const int lane = threadIdx.x & 31;
  const int wave = threadIdx.x >> 5;
  const int grpN = Nout / (16 * NT);               // N-strip groups
  int tile = blockIdx.x * (blockDim.x >> 5) + wave;
  if (tile >= tilesTotal) return;
  const int tm  = tile / grpN;
  const int tg  = tile - tm * grpN;
  const int l15 = lane & 15;
  const int hi  = lane >> 4;

  int rowA = tm * 16 + l15;
  if (rowA >= M) rowA = M - 1;                     // clamped dup load; store guarded
  const bf16_t* pa = A + (size_t)rowA * K + hi * 8;            // A frag: K = {k0+hi*8..+7, +16}
  const bf16_t* pb[NT];
#pragma unroll
  for (int j = 0; j < NT; ++j)                                  // B frag: K = {k0+hi*8..+7, +16}
    pb[j] = Bt + (size_t)(tg * (16 * NT) + j * 16 + l15) * K + hi * 8;

  v8f acc[NT];
#pragma unroll
  for (int j = 0; j < NT; ++j) acc[j] = (v8f){0.f,0.f,0.f,0.f,0.f,0.f,0.f,0.f};

  for (int k0 = 0; k0 < K; k0 += 32) {
    __builtin_prefetch(pa + 64, 0, 1);             // global_prefetch_b8, next-next A chunk
    v8bf a0 = *(const v8bf*)(pa);                  // K = k0 + hi*8 .. +7
    v8bf a1 = *(const v8bf*)(pa + 16);             // K = k0 + 16 + hi*8 .. +7
    v16bf av;
#pragma unroll
    for (int i = 0; i < 8; ++i) { av[i] = a0[i]; av[i + 8] = a1[i]; }
#pragma unroll
    for (int j = 0; j < NT; ++j) {
      v8bf b0 = *(const v8bf*)(pb[j]);             // K = k0 + hi*8 .. +7
      v8bf b1 = *(const v8bf*)(pb[j] + 16);        // K = k0 + 16 + hi*8 .. +7
      v16bf b;
#pragma unroll
      for (int i = 0; i < 8; ++i) { b[i] = b0[i]; b[i + 8] = b1[i]; }
      acc[j] = __builtin_amdgcn_wmma_f32_16x16x32_bf16(false, av, false, b,
                                                       (short)0, acc[j], false, false);
      pb[j] += 32;
    }
    pa += 32;
  }

  // C/D layout: lane<16 -> M=r, lane>=16 -> M=r+8; N = lane&15
  const int row0 = tm * 16 + hi * 8;
#pragma unroll
  for (int j = 0; j < NT; ++j) {
    const int col = tg * (16 * NT) + j * 16 + l15;
#pragma unroll
    for (int r = 0; r < 8; ++r) {
      int row = row0 + r;
      if (row < M) C[(size_t)row * Nout + col] = acc[j][r];
    }
  }
}

// ------------------------------------------------------------- attention math
__global__ void k_attn_scores(const float* __restrict__ h, const float* __restrict__ a_s,
                              const float* __restrict__ a_d, float* __restrict__ es,
                              float* __restrict__ ed, int Nn, int H, int Cc) {
  int t = blockIdx.x * blockDim.x + threadIdx.x;
  if (t >= Nn * H) return;
  int n = t / H, hh = t - n * H;
  const float* hp = h + (size_t)n * H * Cc + (size_t)hh * Cc;
  const float* as = a_s + hh * Cc;
  const float* ad = a_d + hh * Cc;
  float s = 0.f, d = 0.f;
  for (int c = 0; c < Cc; ++c) { float v = hp[c]; s += v * as[c]; d += v * ad[c]; }
  es[t] = s; ed[t] = d;
}

__device__ __forceinline__ void atomic_max_f32(float* addr, float val) {
  unsigned int* ua = (unsigned int*)addr;
  unsigned int old = *ua;
  while (__uint_as_float(old) < val) {
    unsigned int assumed = old;
    old = atomicCAS(ua, assumed, __float_as_uint(val));
    if (old == assumed) break;
  }
}

__device__ __forceinline__ void edge_ends(const long long* __restrict__ ei, int E_, int e,
                                          long long& s, long long& d) {
  if (e < E_) { s = ei[e]; d = ei[(size_t)E_ + e]; }
  else        { s = d = (long long)(e - E_); }          // appended self-loops
}

__global__ void k_edge_logit_max(const long long* __restrict__ ei, int E_, int Nn, int H,
                                 const float* __restrict__ es, const float* __restrict__ ed,
                                 float* __restrict__ elog, float* __restrict__ mb) {
  long long total = (long long)(E_ + Nn) * H;
  long long t = (long long)blockIdx.x * blockDim.x + threadIdx.x;
  if (t >= total) return;
  int e  = (int)(t / H);
  int hh = (int)(t - (long long)e * H);
  long long s, d; edge_ends(ei, E_, e, s, d);
  float v = es[(size_t)s * H + hh] + ed[(size_t)d * H + hh];
  v = v > 0.f ? v : NEG_SLOPE * v;
  elog[t] = v;
  atomic_max_f32(&mb[(size_t)d * H + hh], v);
}

__global__ void k_edge_exp_sum(const long long* __restrict__ ei, int E_, int Nn, int H,
                               const float* __restrict__ mb, float* __restrict__ elog,
                               float* __restrict__ denom) {
  long long total = (long long)(E_ + Nn) * H;
  long long t = (long long)blockIdx.x * blockDim.x + threadIdx.x;
  if (t >= total) return;
  int e  = (int)(t / H);
  int hh = (int)(t - (long long)e * H);
  long long s, d; edge_ends(ei, E_, e, s, d);
  float ex = __expf(elog[t] - mb[(size_t)d * H + hh]);
  elog[t] = ex;
  atomicAdd(&denom[(size_t)d * H + hh], ex);
}

// 4 channels per thread: coalesced float4 gather of h[src], f32 atomic scatter to agg[dst]
__global__ void k_edge_aggregate(const long long* __restrict__ ei, int E_, int Nn, int H, int Cc,
                                 const float* __restrict__ h, const float* __restrict__ elog,
                                 const float* __restrict__ denom, float* __restrict__ agg) {
  const int HC  = H * Cc;
  const int tpe = HC >> 2;
  long long gid  = (long long)blockIdx.x * blockDim.x + threadIdx.x;
  long long edge = gid / tpe;
  if (edge >= (long long)(E_ + Nn)) return;
  int lc = (int)(gid - edge * tpe);
  int c0 = lc << 2;
  int hh = c0 / Cc;
  long long s, d; edge_ends(ei, E_, (int)edge, s, d);
  float alpha = elog[edge * H + hh] / (denom[(size_t)d * H + hh] + 1e-16f);
  const float4 hv = *(const float4*)(h + (size_t)s * HC + c0);
  float* out = agg + (size_t)d * HC + c0;
  atomicAdd(out + 0, hv.x * alpha);
  atomicAdd(out + 1, hv.y * alpha);
  atomicAdd(out + 2, hv.z * alpha);
  atomicAdd(out + 3, hv.w * alpha);
}

// hcat = bf16([relu(agg + bias), x])  -> input of next layer's WMMA GEMM
__global__ void k_finalize_concat(const float* __restrict__ agg, const float* __restrict__ bias,
                                  const float* __restrict__ x, bf16_t* __restrict__ hcat,
                                  int Nn, int HC, int F) {
  size_t total = (size_t)Nn * (HC + F);
  size_t i  = (size_t)blockIdx.x * blockDim.x + threadIdx.x;
  size_t st = (size_t)gridDim.x * blockDim.x;
  for (; i < total; i += st) {
    int n = (int)(i / (HC + F));
    int j = (int)(i - (size_t)n * (HC + F));
    float v;
    if (j < HC) { v = agg[(size_t)n * HC + j] + bias[j]; v = v > 0.f ? v : 0.f; }
    else        { v = x[(size_t)n * F + (j - HC)]; }
    hcat[i] = (bf16_t)v;
  }
}

__global__ void k_final_logsoftmax(const float* __restrict__ agg3, const float* __restrict__ b3,
                                   float* __restrict__ out, int Nn) {
  int n = blockIdx.x * blockDim.x + threadIdx.x;
  if (n >= Nn) return;
  float v[16];
  float mx = -3.402823466e38f;
  for (int c = 0; c < 16; ++c) {
    v[c] = agg3[(size_t)n * 16 + c] + b3[c];
    mx = fmaxf(mx, v[c]);
  }
  float s = 0.f;
  for (int c = 0; c < 16; ++c) s += __expf(v[c] - mx);
  float l = __logf(s);
  for (int c = 0; c < 16; ++c) out[(size_t)n * 16 + c] = v[c] - mx - l;
}

// -------------------------------------------------------------------- driver
static inline int gblk(long long total, int bs) { return (int)((total + bs - 1) / bs); }

extern "C" void kernel_launch(void* const* d_in, const int* in_sizes, int n_in,
                              void* d_out, int out_size, void* d_ws, size_t ws_size,
                              hipStream_t stream) {
  const int F_IN = 256, HID = 64, HEADS = 8, OUT = 16;
  const int HC = HEADS * HID;            // 512
  const int IN2 = HC + F_IN;             // 768
  const int N = in_sizes[0] / F_IN;
  const int E = in_sizes[1] / 2;
  const int E2 = E + N;

  const float*     x   = (const float*)d_in[0];
  const long long* ei  = (const long long*)d_in[1];
  const float *w1 = (const float*)d_in[2],  *a1s = (const float*)d_in[3],
              *a1d = (const float*)d_in[4], *b1  = (const float*)d_in[5];
  const float *w2 = (const float*)d_in[6],  *a2s = (const float*)d_in[7],
              *a2d = (const float*)d_in[8], *b2  = (const float*)d_in[9];
  const float *w3 = (const float*)d_in[10], *a3s = (const float*)d_in[11],
              *a3d = (const float*)d_in[12], *b3 = (const float*)d_in[13];
  float* out = (float*)d_out;

  // carve workspace
  char* base = (char*)d_ws;
  size_t off = 0;
  auto carve = [&](size_t bytes) -> void* {
    void* p = base + off;
    off = (off + bytes + 255) & ~(size_t)255;
    return p;
  };
  bf16_t* xb   = (bf16_t*)carve((size_t)N * F_IN * 2);
  bf16_t* w1t  = (bf16_t*)carve((size_t)HC * F_IN * 2);     // [512,256] K-major
  bf16_t* w2t  = (bf16_t*)carve((size_t)HC * IN2 * 2);      // [512,768]
  bf16_t* w3t  = (bf16_t*)carve((size_t)OUT * IN2 * 2);     // [16,768]
  float*  h    = (float*)carve((size_t)N * HC * 4);         // projections (reused L1/L2/L3)
  float*  agg  = (float*)carve((size_t)N * HC * 4);         // aggregation  (reused)
  bf16_t* hcat = (bf16_t*)carve((size_t)N * IN2 * 2);       // concat input (reused)
  float*  es   = (float*)carve((size_t)N * HEADS * 4);
  float*  ed   = (float*)carve((size_t)N * HEADS * 4);
  float*  mb   = (float*)carve((size_t)N * HEADS * 4);
  float*  dn   = (float*)carve((size_t)N * HEADS * 4);
  float*  elog = (float*)carve((size_t)E2 * HEADS * 4);
  (void)ws_size; (void)n_in; (void)out_size;

  const int BS = 256;
  const float NEG_INF = -3.402823466e38f;

  // ---- static preprocessing (bf16 casts + weight transposes)
  k_cast_bf16<<<gblk((long long)N * F_IN, BS), BS, 0, stream>>>(x, xb, (size_t)N * F_IN);
  k_transpose_bf16<<<gblk((long long)F_IN * HC, BS), BS, 0, stream>>>(w1, w1t, F_IN, HC);
  k_transpose_bf16<<<gblk((long long)IN2 * HC, BS), BS, 0, stream>>>(w2, w2t, IN2, HC);
  k_transpose_bf16<<<gblk((long long)IN2 * OUT, BS), BS, 0, stream>>>(w3, w3t, IN2, OUT);

  auto run_edge_layer = [&](const float* hl, const float* as_, const float* ad_,
                            int H, int Cc) {
    k_attn_scores<<<gblk((long long)N * H, BS), BS, 0, stream>>>(hl, as_, ad_, es, ed, N, H, Cc);
    k_fill_f32<<<gblk((long long)N * H, BS), BS, 0, stream>>>(mb, NEG_INF, (size_t)N * H);
    k_fill_f32<<<gblk((long long)N * H, BS), BS, 0, stream>>>(dn, 0.f, (size_t)N * H);
    k_fill_f32<<<gblk((long long)N * H * Cc, BS), BS, 0, stream>>>(agg, 0.f, (size_t)N * H * Cc);
    k_edge_logit_max<<<gblk((long long)E2 * H, BS), BS, 0, stream>>>(ei, E, N, H, es, ed, elog, mb);
    k_edge_exp_sum<<<gblk((long long)E2 * H, BS), BS, 0, stream>>>(ei, E, N, H, mb, elog, dn);
    k_edge_aggregate<<<gblk((long long)E2 * (H * Cc / 4), BS), BS, 0, stream>>>(
        ei, E, N, H, Cc, hl, elog, dn, agg);
  };

  const int tilesM = (N + 15) / 16;

  // ---- layer 1: h = xb @ w1 ; GAT aggregate ; hcat = [relu(agg+b1), x]
  {
    int tiles = tilesM * (HC / 64);                       // 16x64 strips (NT=4)
    k_gemm_bf16_wmma<4><<<gblk(tiles, 4), 128, 0, stream>>>(xb, w1t, h, N, F_IN, HC, tiles);
    run_edge_layer(h, a1s, a1d, HEADS, HID);
    k_finalize_concat<<<gblk((long long)N * IN2, BS), BS, 0, stream>>>(agg, b1, x, hcat, N, HC, F_IN);
  }
  // ---- layer 2: h = hcat @ w2 ; GAT aggregate ; hcat = [relu(agg+b2), x]
  {
    int tiles = tilesM * (HC / 64);
    k_gemm_bf16_wmma<4><<<gblk(tiles, 4), 128, 0, stream>>>(hcat, w2t, h, N, IN2, HC, tiles);
    run_edge_layer(h, a2s, a2d, HEADS, HID);
    k_finalize_concat<<<gblk((long long)N * IN2, BS), BS, 0, stream>>>(agg, b2, x, hcat, N, HC, F_IN);
  }
  // ---- layer 3 (heads=1, C=16): h = hcat @ w3 ; aggregate ; log_softmax
  {
    int tiles = tilesM * (OUT / 16);                      // single 16x16 tile (NT=1)
    k_gemm_bf16_wmma<1><<<gblk(tiles, 4), 128, 0, stream>>>(hcat, w3t, h, N, IN2, OUT, tiles);
    run_edge_layer(h, a3s, a3d, 1, OUT);
    k_final_logsoftmax<<<gblk(N, BS), BS, 0, stream>>>(agg, b3, out, N);
  }
}
